// MultiAttention_32796370272586
// MI455X (gfx1250) — matrix operand order repaired and checked by reading, not compiled, not run on hardware
//
#include <hip/hip_runtime.h>
#include <hip/hip_bf16.h>

// ---------------- CDNA5 WMMA types ----------------
typedef __attribute__((ext_vector_type(16))) __bf16 v16bf;
typedef __attribute__((ext_vector_type(8)))  __bf16 v8bf;
typedef __attribute__((ext_vector_type(4)))  __bf16 v4bf;
typedef __attribute__((ext_vector_type(8)))  float  v8f;
typedef __attribute__((ext_vector_type(4)))  float  v4f;

#define TRAN_DIM 768
#define N_HEADS  8
#define HEAD_DIM 96
#define INNER    768
#define SEQ      1024
#define BATCH    16
#define ROWS     (BATCH * SEQ)   // 16384

// ------- async global->LDS via inline CDNA5 asm (ASYNCcnt-tracked) ---------
// ISA 08_async_tensor: GLOBAL_LOAD_ASYNC_TO_LDS_B128, GV mode:
//   VDST = per-lane LDS byte address, VADDR = 64-bit global address.
static __device__ __forceinline__ void asyncCopy16B(const __bf16* g, __bf16* l) {
#if defined(__gfx1250__)
  unsigned lofs =
      (unsigned)(size_t)(__attribute__((address_space(3))) const __bf16*)l;
  unsigned long long ga = (unsigned long long)g;
  asm volatile("global_load_async_to_lds_b128 %0, %1, off"
               :: "v"(lofs), "v"(ga) : "memory");
#else
  *(v8bf*)l = *(const v8bf*)g;
#endif
}

static __device__ __forceinline__ void waitAsync() {
#if defined(__gfx1250__)
  asm volatile("s_wait_asynccnt 0x0" ::: "memory");
#endif
}

// A-fragment (16x32 bf16, ISA 7.12.2): lane row = lane&15, half h = lane>>4.
// elements 0..7 -> K = h*8 + e ; elements 8..15 -> K = 16 + h*8 + (e-8)
static __device__ __forceinline__ v16bf ldsFragA(const __bf16* base, int row,
                                                 int ld, int k0, int h) {
  const __bf16* p = base + row * ld + k0;
  v8bf lo = *(const v8bf*)(p + h * 8);
  v8bf hi = *(const v8bf*)(p + 16 + h * 8);
  v16bf r;
#pragma unroll
  for (int i = 0; i < 8; ++i) { r[i] = lo[i]; r[i + 8] = hi[i]; }
  return r;
}

// B-fragment (32x16 bf16): lane col = lane&15; elements 0..7 -> K = h*8 + e ; elements 8..15 -> K = 16 + h*8 + (e-8).
static __device__ __forceinline__ v16bf ldsFragB(const __bf16* base, int col,
                                                 int ld, int k0, int h) {
  const __bf16* p = base + col * ld + k0;
  v8bf lo = *(const v8bf*)(p + h * 8);
  v8bf hi = *(const v8bf*)(p + 16 + h * 8);
  v16bf r;
#pragma unroll
  for (int i = 0; i < 8; ++i) { r[i] = lo[i]; r[i + 8] = hi[i]; }
  return r;
}

// ---------------- GEMM: C[M,N] = A[M,K] @ B[K,N] + bias ----------------
// 256 threads / 8 waves; 256x64 block tile, 32x64 per wave, BK=32,
// register-staged software pipeline on the global->LDS path.
// MODE 0: A f32, out scattered bf16 into qkv ws [3][B][H][N][D]
// MODE 1: A bf16, out plain f32 [M,N]
template <int MODE>
__global__ __launch_bounds__(256) void gemm_wmma_kernel(
    const void* __restrict__ Ap, const float* __restrict__ B,
    const float* __restrict__ bias, void* __restrict__ Out,
    int M, int N, int K) {
  __shared__ alignas(16) __bf16 Al[256 * 32];
  __shared__ alignas(16) __bf16 Bl[64 * 32];

  const int tid  = threadIdx.x;
  const int wave = tid >> 5;
  const int lane = tid & 31;
  const int ln   = lane & 15;
  const int h    = lane >> 4;
  const int mBase = blockIdx.y * 256;
  const int nBase = blockIdx.x * 64;
  (void)M;

  v8f acc[8] = {};
  v4f  a4[8];        // MODE 0 staging
  v8bf a8[4];        // MODE 1 staging
  float breg[8];

  auto loadRegs = [&](int k0) {
    if (MODE == 0) {
      const float* A = (const float*)Ap;
#pragma unroll
      for (int i = 0; i < 8; ++i) {
        int idx = i * 256 + tid;
        int r = idx >> 3, c4 = (idx & 7) * 4;
        a4[i] = *(const v4f*)(A + (size_t)(mBase + r) * K + k0 + c4);
      }
    } else {
      const __bf16* A = (const __bf16*)Ap;
#pragma unroll
      for (int i = 0; i < 4; ++i) {
        int idx = i * 256 + tid;
        int r = idx >> 2, c8 = (idx & 3) * 8;
        a8[i] = *(const v8bf*)(A + (size_t)(mBase + r) * K + k0 + c8);
      }
    }
#pragma unroll
    for (int i = 0; i < 8; ++i) {
      int idx = i * 256 + tid;
      int kk = idx >> 6, nn = idx & 63;
      breg[i] = B[(size_t)(k0 + kk) * N + nBase + nn];
    }
  };

  auto storeLds = [&]() {
    if (MODE == 0) {
#pragma unroll
      for (int i = 0; i < 8; ++i) {
        int idx = i * 256 + tid;
        int r = idx >> 3, c4 = (idx & 7) * 4;
        v4bf y;
#pragma unroll
        for (int j = 0; j < 4; ++j) y[j] = (__bf16)a4[i][j];
        *(v4bf*)(Al + r * 32 + c4) = y;
      }
    } else {
#pragma unroll
      for (int i = 0; i < 4; ++i) {
        int idx = i * 256 + tid;
        int r = idx >> 2, c8 = (idx & 3) * 8;
        *(v8bf*)(Al + r * 32 + c8) = a8[i];
      }
    }
#pragma unroll
    for (int i = 0; i < 8; ++i) {          // B transposed -> Bl[n][k]
      int idx = i * 256 + tid;
      int kk = idx >> 6, nn = idx & 63;
      Bl[nn * 32 + kk] = (__bf16)breg[i];
    }
  };

  loadRegs(0);
  for (int k0 = 0; k0 < K; k0 += 32) {
    __syncthreads();                 // previous readers done with LDS
    storeLds();
    __syncthreads();
    if (k0 + 32 < K) {
      loadRegs(k0 + 32);             // overlaps with WMMA stream below
      __builtin_prefetch(B + (size_t)(k0 + 64) * N + nBase + (tid & 63), 0, 1);
    }
    v16bf A0 = ldsFragA(Al, wave * 32 + ln,      32, 0, h);
    v16bf A1 = ldsFragA(Al, wave * 32 + 16 + ln, 32, 0, h);
#pragma unroll
    for (int t = 0; t < 4; ++t) {
      v16bf b = ldsFragB(Bl, t * 16 + ln, 32, 0, h);
      acc[t]     = __builtin_amdgcn_wmma_f32_16x16x32_bf16(
          false, A0, false, b, (short)0, acc[t], false, false);
      acc[4 + t] = __builtin_amdgcn_wmma_f32_16x16x32_bf16(
          false, A1, false, b, (short)0, acc[4 + t], false, false);
    }
  }

  // C layout: lane holds col n = ln, rows m = j + 8*h
#pragma unroll
  for (int r = 0; r < 2; ++r) {
#pragma unroll
    for (int t = 0; t < 4; ++t) {
      int col = nBase + t * 16 + ln;
      float bv = bias[col];
#pragma unroll
      for (int j = 0; j < 8; ++j) {
        int m = mBase + wave * 32 + r * 16 + j + 8 * h;
        float vv = acc[r * 4 + t][j] + bv;
        if (MODE == 0) {
          int which = col / INNER;
          int inner = col - which * INNER;
          int hh = inner / HEAD_DIM;
          int dd = inner - hh * HEAD_DIM;
          int b_ = m >> 10, n_ = m & (SEQ - 1);
          __bf16* outp = (__bf16*)Out;
          size_t off = (((size_t)which * BATCH + b_) * N_HEADS + hh) *
                           ((size_t)SEQ * HEAD_DIM) +
                       (size_t)n_ * HEAD_DIM + dd;
          outp[off] = (__bf16)vv;
        } else {
          float* outp = (float*)Out;
          outp[(size_t)m * N + col] = vv;
        }
      }
    }
  }
}

// ---------------- Flash attention per (b,h), 64-query x 64-key tiles -------
// 128 threads / 4 waves, 16 q-rows per wave, double-buffered K/V^T in LDS
// (K tile via async-to-LDS inline asm), Q fragments cached in registers.
__global__ __launch_bounds__(128) void attn_wmma_kernel(
    const __bf16* __restrict__ q, const __bf16* __restrict__ k,
    const __bf16* __restrict__ v, __bf16* __restrict__ o) {
  constexpr int KT = 64;
  constexpr int KV_ELEMS = KT * 96 + 96 * KT;        // 12288 per buffer
  __shared__ alignas(16) __bf16 smem[2 * KV_ELEMS + 4 * 16 * KT]; // 56 KB

  const int tid  = threadIdx.x;
  const int wave = tid >> 5;
  const int lane = tid & 31;
  const int ln   = lane & 15;
  const int h    = lane >> 4;
  __bf16* Pl = smem + 2 * KV_ELEMS + wave * 16 * KT;
  __bf16* Ql = smem;                                  // staging aliases buf 0

  const int bh = blockIdx.y;                          // b*8 + head
  const int q0 = blockIdx.x * 64;
  const size_t base = (size_t)bh * SEQ * HEAD_DIM;
  const float scale = 0.10206207261596575f;           // 1/sqrt(96)

  // ---- stage Q tile, capture A-fragments in registers, then recycle LDS ---
#pragma unroll
  for (int i = 0; i < 6; ++i) {
    int idx = i * 128 + tid;
    int r = idx / 12, c8 = (idx % 12) * 8;
    *(v8bf*)(Ql + r * 96 + c8) =
        *(const v8bf*)(q + base + (size_t)(q0 + r) * 96 + c8);
  }
  __syncthreads();
  v16bf qa[3];
#pragma unroll
  for (int d0 = 0; d0 < 96; d0 += 32)
    qa[d0 >> 5] = ldsFragA(Ql, wave * 16 + ln, 96, d0, h);
  __syncthreads();

  auto loadKV = [&](int kt, __bf16* Kl, __bf16* Vt) {
#pragma unroll
    for (int i = 0; i < 6; ++i) {           // K tile [64][96] straight copy
      int idx = i * 128 + tid;
      int r = idx / 12, c8 = (idx % 12) * 8;
      asyncCopy16B(k + base + (size_t)(kt + r) * 96 + c8, Kl + r * 96 + c8);
    }
#pragma unroll
    for (int i = 0; i < 6; ++i) {           // V tile transposed -> Vt[96][64]
      int idx = i * 128 + tid;
      int r = idx / 12, c8 = (idx % 12) * 8;
      v8bf val = *(const v8bf*)(v + base + (size_t)(kt + r) * 96 + c8);
#pragma unroll
      for (int j = 0; j < 8; ++j) Vt[(c8 + j) * KT + r] = val[j];
    }
  };

  float mrow[8], lrow[8];
  v8f o_acc[6] = {};
#pragma unroll
  for (int j = 0; j < 8; ++j) { mrow[j] = -1e30f; lrow[j] = 0.f; }

  loadKV(0, smem, smem + KT * 96);          // prologue into buffer 0

  for (int it = 0; it < SEQ / KT; ++it) {
    const int curOfs = (it & 1) * KV_ELEMS;       // buffer select by offset
    const int nxtOfs = ((it + 1) & 1) * KV_ELEMS; // (no LDS pointer arrays)
    const __bf16* Kc = smem + curOfs;
    const __bf16* Vc = smem + curOfs + KT * 96;
    waitAsync();
    __syncthreads();                        // buffer `cur` fully resident
    if (it + 1 < SEQ / KT)                  // prefetch next tile (other buf)
      loadKV((it + 1) * KT, smem + nxtOfs, smem + nxtOfs + KT * 96);

    // ---- S = Q * K^T over d = 96 (3 WMMA k-steps), 4 key-column tiles ----
    v8f s[4] = {};
#pragma unroll
    for (int d0 = 0; d0 < 96; d0 += 32) {
      v16bf a = qa[d0 >> 5];
#pragma unroll
      for (int t = 0; t < 4; ++t) {
        v16bf b = ldsFragB(Kc, t * 16 + ln, 96, d0, h);
        s[t] = __builtin_amdgcn_wmma_f32_16x16x32_bf16(
            false, a, false, b, (short)0, s[t], false, false);
      }
    }

    // ---- online softmax: rows m = j + 8*h, reduce across 16 column-lanes --
#pragma unroll
    for (int j = 0; j < 8; ++j) {
      float a0 = s[0][j] * scale, a1 = s[1][j] * scale;
      float a2 = s[2][j] * scale, a3 = s[3][j] * scale;
      float mx = fmaxf(fmaxf(a0, a1), fmaxf(a2, a3));
#pragma unroll
      for (int d = 1; d < 16; d <<= 1) mx = fmaxf(mx, __shfl_xor(mx, d, 32));
      float mn = fmaxf(mrow[j], mx);
      float corr = __expf(mrow[j] - mn);
      mrow[j] = mn;
      float p0 = __expf(a0 - mn), p1 = __expf(a1 - mn);
      float p2 = __expf(a2 - mn), p3 = __expf(a3 - mn);
      float ps = (p0 + p1) + (p2 + p3);
#pragma unroll
      for (int d = 1; d < 16; d <<= 1) ps += __shfl_xor(ps, d, 32);
      lrow[j] = lrow[j] * corr + ps;
#pragma unroll
      for (int u = 0; u < 6; ++u) o_acc[u][j] *= corr;
      __bf16* pr = Pl + (j + 8 * h) * KT + ln;
      pr[0]  = (__bf16)p0;
      pr[16] = (__bf16)p1;
      pr[32] = (__bf16)p2;
      pr[48] = (__bf16)p3;
    }
    // DS ops are in-order within a wave: P stores are visible to loads below.

    // ---- O += P * V : A = P (16x64, 2 k-steps), B = V^T columns -----------
#pragma unroll
    for (int kh = 0; kh < 2; ++kh) {
      v16bf pa = ldsFragA(Pl, ln, KT, kh * 32, h);
#pragma unroll
      for (int u = 0; u < 6; ++u) {
        v16bf vb = ldsFragB(Vc, u * 16 + ln, KT, kh * 32, h);
        o_acc[u] = __builtin_amdgcn_wmma_f32_16x16x32_bf16(
            false, pa, false, vb, (short)0, o_acc[u], false, false);
      }
    }
  }

  // ---- epilogue: O /= l, write bf16 to attn ws [B][N][INNER] --------------
  const int b_  = bh >> 3;
  const int hd0 = (bh & 7) * HEAD_DIM;
#pragma unroll
  for (int u = 0; u < 6; ++u) {
#pragma unroll
    for (int j = 0; j < 8; ++j) {
      int row = q0 + wave * 16 + j + 8 * h;
      int d   = u * 16 + ln;
      float val = o_acc[u][j] / lrow[j];
      o[((size_t)b_ * SEQ + row) * INNER + hd0 + d] = (__bf16)val;
    }
  }
}

// ---------------- launch ----------------
extern "C" void kernel_launch(void* const* d_in, const int* in_sizes, int n_in,
                              void* d_out, int out_size, void* d_ws,
                              size_t ws_size, hipStream_t stream) {
  (void)in_sizes; (void)n_in; (void)out_size; (void)ws_size;
  const float* x     = (const float*)d_in[0];
  const float* W_qkv = (const float*)d_in[1];
  const float* b_qkv = (const float*)d_in[2];
  const float* W_out = (const float*)d_in[3];
  const float* b_out = (const float*)d_in[4];

  const size_t headElems = (size_t)BATCH * N_HEADS * SEQ * HEAD_DIM; // 12.58M
  __bf16* qkv  = (__bf16*)d_ws;                 // [3][B][H][N][D] bf16
  __bf16* attn = qkv + 3 * headElems;           // [B][N][INNER]   bf16

  dim3 g1((3 * INNER) / 64, ROWS / 256);
  gemm_wmma_kernel<0><<<g1, 256, 0, stream>>>(x, W_qkv, b_qkv, qkv,
                                              ROWS, 3 * INNER, TRAN_DIM);

  const __bf16* qp = qkv;
  const __bf16* kp = qkv + headElems;
  const __bf16* vp = qkv + 2 * headElems;
  dim3 g2(SEQ / 64, BATCH * N_HEADS);
  attn_wmma_kernel<<<g2, 128, 0, stream>>>(qp, kp, vp, attn);

  dim3 g3(TRAN_DIM / 64, ROWS / 256);
  gemm_wmma_kernel<1><<<g3, 256, 0, stream>>>(attn, W_out, b_out, d_out,
                                              ROWS, TRAN_DIM, INNER);
}
